// LinearTemporalDiffusionTransformerDecoderLayer_64046552318271
// MI455X (gfx1250) — matrix operand order repaired and checked by reading, not compiled, not run on hardware
//
#include <hip/hip_runtime.h>
#include <hip/hip_bf16.h>

// ---------------------------------------------------------------------------
// LinearTemporalDiffusionTransformerDecoderLayer for MI455X (gfx1250).
//  - all 8 large GEMMs (~195/200 GFLOP): bf16 v_wmma_f32_16x16x32_bf16,
//    tiles staged with GLOBAL_LOAD_ASYNC_TO_LDS_B128 (ASYNCcnt) when full.
//  - self-attention (~15 GFLOP): WMMA flash-attention (QK^T and P.V on the
//    matrix pipe, online softmax via wave32 shuffle reductions).
//  - LN / tiny linear-attention einsums / softmaxes: fp32 VALU.
// Shapes: T=256 B=64 N=77 D=512 S=334 H=8 dh=64 FF_SA=1024 FF=2048.
// ---------------------------------------------------------------------------

typedef __bf16 bf16;
typedef __attribute__((ext_vector_type(16))) __bf16 v16bf;
typedef __attribute__((ext_vector_type(8)))  __bf16 v8bf;
typedef __attribute__((ext_vector_type(8)))  float  v8f;

#define Tn   256
#define Bn   64
#define Nn   77
#define Dn   512
#define Sn   334
#define Hn   8
#define DHn  64

static constexpr size_t SBD = (size_t)Sn * Bn * Dn;   // 10,944,512
static constexpr size_t BTD = (size_t)Bn * Tn * Dn;   //  8,388,608
static constexpr size_t BND = (size_t)Bn * Nn * Dn;   //  2,523,136
static constexpr int    SB  = Sn * Bn;                // 21,376
static constexpr int    BT  = Bn * Tn;                // 16,384
static constexpr int    BNr = Bn * Nn;                //  4,928

// -------------------------- async LDS helpers ------------------------------

__device__ inline void async_load_b128(void* lds_dst, const void* gsrc) {
  unsigned l = (unsigned)(size_t)lds_dst;                 // low 32 bits = LDS offset
  unsigned long long g = (unsigned long long)(size_t)gsrc;
  asm volatile("global_load_async_to_lds_b128 %0, %1, off"
               :: "v"(l), "v"(g) : "memory");
}
__device__ inline void async_wait0() {
  asm volatile("s_wait_asynccnt 0" ::: "memory");
}

// ------------------------------ elementwise --------------------------------

__global__ void k_f2b(const float* __restrict__ in, bf16* __restrict__ out, size_t n) {
  size_t i = (size_t)blockIdx.x * 256 + threadIdx.x;
  if (i < n) out[i] = (bf16)in[i];
}

__global__ void k_silu_b(const float* __restrict__ in, bf16* __restrict__ out, size_t n) {
  size_t i = (size_t)blockIdx.x * 256 + threadIdx.x;
  if (i < n) { float v = in[i]; out[i] = (bf16)(v / (1.f + __expf(-v))); }
}

__global__ void k_build_src(const float* __restrict__ x, const float* __restrict__ xf,
                            const float* __restrict__ emb,
                            float* __restrict__ srcF, bf16* __restrict__ srcB) {
  size_t i = (size_t)blockIdx.x * 256 + threadIdx.x;
  if (i >= SBD) return;
  size_t s = i / ((size_t)Bn * Dn), rem = i % ((size_t)Bn * Dn);
  float v;
  if (s < Tn)            v = x[i];
  else if (s < Tn + Nn)  v = xf[(s - Tn) * (size_t)Bn * Dn + rem];
  else                   v = emb[rem];
  srcF[i] = v;
  srcB[i] = (bf16)v;
}

// out[r*512+c] = in[ ((r/P)*sQ + (r%P)*sR) * 512 + c ]
__global__ void k_permute(const float* __restrict__ in, float* __restrict__ out,
                          int M, int P, int sQ, int sR) {
  size_t i = (size_t)blockIdx.x * 256 + threadIdx.x;
  if (i >= (size_t)M * Dn) return;
  int r = (int)(i >> 9), c = (int)(i & 511);
  size_t ir = (size_t)(r / P) * sQ + (size_t)(r % P) * sR;
  out[i] = in[ir * Dn + c];
}

// ------------------------------ WMMA GEMM ----------------------------------
// C[M,N] = act(A[M,K] @ W[N,K]^T + bias[N]) + resid[M,N]
// A,W bf16 row-major. N,K multiples of 128/64. act:0 none,1 relu,2 gelu,3 silu.
// Full M-tiles stage A/W via GLOBAL_LOAD_ASYNC_TO_LDS_B128.

#define GBM 128
#define GBN 128
#define GBK 64

__global__ __launch_bounds__(256) void k_gemm(
    const bf16* __restrict__ A, const bf16* __restrict__ W,
    const float* __restrict__ bias, const float* __restrict__ resid,
    float* __restrict__ outF, bf16* __restrict__ outB,
    int M, int N, int K, int act)
{
  __shared__ bf16 sA[GBM * GBK];
  __shared__ bf16 sB[GBN * GBK];

  const int tid  = threadIdx.x;
  const int lane = tid & 31;
  const int wave = tid >> 5;
  const int wm   = wave & 3;        // 4 row-groups of 32
  const int wn   = wave >> 2;       // 2 col-groups of 64
  const int half = lane >> 4;
  const int l16  = lane & 15;
  const int bm   = blockIdx.y * GBM;
  const int bn   = blockIdx.x * GBN;
  const bool fullM = (bm + GBM) <= M;

  v8f acc[2][4];
#pragma unroll
  for (int i = 0; i < 2; ++i)
#pragma unroll
    for (int j = 0; j < 4; ++j) acc[i][j] = 0;

  for (int k0 = 0; k0 < K; k0 += GBK) {
    if (fullM) {
      // CDNA5 async path: DMA 16B per lane straight into LDS, ASYNCcnt-tracked.
#pragma unroll
      for (int r = 0; r < 4; ++r) {
        int row = (tid >> 3) + r * 32;
        int col = (tid & 7) * 8;
        async_load_b128(sA + row * GBK + col, A + (size_t)(bm + row) * K + k0 + col);
        async_load_b128(sB + row * GBK + col, W + (size_t)(bn + row) * K + k0 + col);
      }
      async_wait0();
    } else {
#pragma unroll
      for (int r = 0; r < 4; ++r) {
        int row = (tid >> 3) + r * 32;
        int col = (tid & 7) * 8;
        uint4 av = make_uint4(0, 0, 0, 0);
        int grow = bm + row;
        if (grow < M) av = *(const uint4*)(A + (size_t)grow * K + k0 + col);
        *(uint4*)(sA + row * GBK + col) = av;
        uint4 wv = *(const uint4*)(W + (size_t)(bn + row) * K + k0 + col);
        *(uint4*)(sB + row * GBK + col) = wv;
      }
    }
    __syncthreads();

#pragma unroll
    for (int kk = 0; kk < GBK; kk += 32) {
      v16bf af[2], bfr[4];
#pragma unroll
      for (int i = 0; i < 2; ++i) {
        // A 16x32 fragment in this code's contiguous k order (not AMD's table order): lane half selects K 0..15 vs 16..31 (matches B)
        const bf16* base = sA + (wm * 32 + i * 16 + l16) * GBK + kk;
        v8bf lo = *(const v8bf*)(base + half * 16);
        v8bf hi = *(const v8bf*)(base + half * 16 + 8);
        af[i] = __builtin_shufflevector(lo, hi, 0, 1, 2, 3, 4, 5, 6, 7,
                                                8, 9, 10, 11, 12, 13, 14, 15);
      }
#pragma unroll
      for (int j = 0; j < 4; ++j) {
        // B fragment in this code's contiguous k order (same as A): lanes 0-15 K=0..15, lanes 16-31 K=16..31
        bfr[j] = *(const v16bf*)(sB + (wn * 64 + j * 16 + l16) * GBK + kk + half * 16);
      }
#pragma unroll
      for (int i = 0; i < 2; ++i)
#pragma unroll
        for (int j = 0; j < 4; ++j)
          acc[i][j] = __builtin_amdgcn_wmma_f32_16x16x32_bf16(
              false, af[i], false, bfr[j], (short)0, acc[i][j], false, false);
    }
    __syncthreads();
  }

  // epilogue: bias -> activation -> residual -> store f32/bf16
#pragma unroll
  for (int i = 0; i < 2; ++i) {
#pragma unroll
    for (int j = 0; j < 4; ++j) {
      int col  = bn + wn * 64 + j * 16 + l16;
      float bv = bias ? bias[col] : 0.f;
#pragma unroll
      for (int r = 0; r < 8; ++r) {
        int row = bm + wm * 32 + i * 16 + half * 8 + r;
        if (row < M) {
          float v = acc[i][j][r] + bv;
          if (act == 1)      v = fmaxf(v, 0.f);
          else if (act == 2) v = 0.5f * v * (1.f + erff(v * 0.70710678f));
          else if (act == 3) v = v / (1.f + __expf(-v));
          if (resid) v += resid[(size_t)row * N + col];
          if (outF)  outF[(size_t)row * N + col] = v;
          if (outB)  outB[(size_t)row * N + col] = (bf16)v;
        }
      }
    }
  }
}

// ------------------------ WMMA flash self-attention ------------------------
// grid (qchunk=3, b*h=512); block 256 = 8 waves; wave owns a 16-row Q tile.
// Per 32-key chunk: S = QK^T (4 wmma), online softmax on accumulator layout,
// P restaged via per-wave LDS patch, O += P.V (4 wmma).
__global__ __launch_bounds__(256) void k_attn(const float* __restrict__ qkv,
                                              bf16* __restrict__ o) {
  const int b  = blockIdx.y >> 3;
  const int h  = blockIdx.y & 7;
  const int q0 = blockIdx.x * 128;

  __shared__ bf16 sQ[128 * 64];     // Q chunk (row-major), 16KB
  __shared__ bf16 sK[32 * 64];      // K chunk [key][d], 4KB
  __shared__ bf16 sVt[64 * 32];     // V chunk transposed [d][key], 4KB
  __shared__ bf16 sP[8][16 * 32];   // per-wave P tile staging, 8KB

  const int tid  = threadIdx.x;
  const int lane = tid & 31;
  const int wave = tid >> 5;
  const int half = lane >> 4;
  const int l16  = lane & 15;

  // stage Q chunk (scaled by dh^-0.5), clamp OOB rows to a valid row
#pragma unroll
  for (int r = 0; r < 32; ++r) {
    int idx = tid + r * 256;
    int row = idx >> 6, d = idx & 63;
    int s = q0 + row; if (s >= Sn) s = Sn - 1;
    sQ[idx] = (bf16)(qkv[((size_t)s * Bn + b) * 1536 + h * DHn + d] * 0.125f);
  }
  __syncthreads();

  // per-wave Q fragments (A 16x32), rows = q0 + wave*16 + l16
  v16bf qf[2];
#pragma unroll
  for (int kk = 0; kk < 2; ++kk) {
    const bf16* base = sQ + (wave * 16 + l16) * 64 + kk * 32;
    v8bf lo = *(const v8bf*)(base + half * 16);
    v8bf hi = *(const v8bf*)(base + half * 16 + 8);
    qf[kk] = __builtin_shufflevector(lo, hi, 0, 1, 2, 3, 4, 5, 6, 7,
                                             8, 9, 10, 11, 12, 13, 14, 15);
  }
  __syncthreads();   // sQ no longer needed coherently; keep simple

  float m_row[8], l_row[8];
#pragma unroll
  for (int r = 0; r < 8; ++r) { m_row[r] = -1e30f; l_row[r] = 0.f; }
  v8f oacc[4];
#pragma unroll
  for (int j = 0; j < 4; ++j) oacc[j] = 0;

  for (int kc = 0; kc < Sn; kc += 32) {
    // stage K chunk and transposed V chunk
#pragma unroll
    for (int r = 0; r < 8; ++r) {
      int idx = tid + r * 256;
      int j = idx >> 6, d = idx & 63;
      int t = kc + j;
      float kv = 0.f, vv = 0.f;
      if (t < Sn) {
        size_t base = ((size_t)t * Bn + b) * 1536 + h * DHn + d;
        kv = qkv[base + 512];
        vv = qkv[base + 1024];
      }
      sK[idx] = (bf16)kv;
      sVt[d * 32 + j] = (bf16)vv;
    }
    __syncthreads();

    // S tiles: two 16-key subtiles, each = Q(16x64) . K_sub(16x64)^T
    v8f st[2];
#pragma unroll
    for (int jt = 0; jt < 2; ++jt) {
      v8f z = 0;
      v16bf kf0 = *(const v16bf*)(sK + (jt * 16 + l16) * 64 + half * 16);
      v16bf kf1 = *(const v16bf*)(sK + (jt * 16 + l16) * 64 + 32 + half * 16);
      z = __builtin_amdgcn_wmma_f32_16x16x32_bf16(false, qf[0], false, kf0,
                                                  (short)0, z, false, false);
      z = __builtin_amdgcn_wmma_f32_16x16x32_bf16(false, qf[1], false, kf1,
                                                  (short)0, z, false, false);
      if (kc + jt * 16 + l16 >= Sn) {     // mask padded keys (this lane's column)
#pragma unroll
        for (int r = 0; r < 8; ++r) z[r] = -1e30f;
      }
      st[jt] = z;
    }

    // online softmax per row (reduce over 16 lanes holding this row's keys)
    float corr_row[8];
#pragma unroll
    for (int r = 0; r < 8; ++r) {
      float cm = fmaxf(st[0][r], st[1][r]);
#pragma unroll
      for (int msk = 1; msk <= 8; msk <<= 1) cm = fmaxf(cm, __shfl_xor(cm, msk, 32));
      float nm = fmaxf(m_row[r], cm);
      float corr = __expf(m_row[r] - nm);
      float p0 = __expf(st[0][r] - nm);
      float p1 = __expf(st[1][r] - nm);
      float ps = p0 + p1;
#pragma unroll
      for (int msk = 1; msk <= 8; msk <<= 1) ps += __shfl_xor(ps, msk, 32);
      l_row[r] = l_row[r] * corr + ps;
      m_row[r] = nm;
      corr_row[r] = corr;
      st[0][r] = p0;
      st[1][r] = p1;
    }
#pragma unroll
    for (int j = 0; j < 4; ++j)
#pragma unroll
      for (int r = 0; r < 8; ++r) oacc[j][r] *= corr_row[r];

    // restage P (acc layout: lane=key, vgpr=row) into A-fragment layout via LDS
    bf16* pw = sP[wave];
#pragma unroll
    for (int jt = 0; jt < 2; ++jt)
#pragma unroll
      for (int r = 0; r < 8; ++r)
        pw[(half * 8 + r) * 32 + jt * 16 + l16] = (bf16)st[jt][r];
    v8bf plo = *(const v8bf*)(pw + l16 * 32 + half * 16);
    v8bf phi = *(const v8bf*)(pw + l16 * 32 + half * 16 + 8);
    v16bf pf = __builtin_shufflevector(plo, phi, 0, 1, 2, 3, 4, 5, 6, 7,
                                                 8, 9, 10, 11, 12, 13, 14, 15);

    // O += P(16x32) . V(32x64), B-fragments from transposed V chunk
#pragma unroll
    for (int j = 0; j < 4; ++j) {
      v16bf vf = *(const v16bf*)(sVt + (j * 16 + l16) * 32 + half * 16);
      oacc[j] = __builtin_amdgcn_wmma_f32_16x16x32_bf16(false, pf, false, vf,
                                                        (short)0, oacc[j], false, false);
    }
    __syncthreads();
  }

  // store O / l   (acc layout: lane=col within 16-group, row = half*8+r)
#pragma unroll
  for (int r = 0; r < 8; ++r) {
    int qrow = q0 + wave * 16 + half * 8 + r;
    if (qrow < Sn) {
      float inv = 1.f / l_row[r];
      size_t ob = ((size_t)qrow * Bn + b) * Dn + h * DHn;
#pragma unroll
      for (int j = 0; j < 4; ++j) o[ob + j * 16 + l16] = (bf16)(oacc[j][r] * inv);
    }
  }
}

// ------------------------------ LayerNorm ----------------------------------
// wave-per-row LN over 512; input row mapping: in_row = (r/P)*sQ + (r%P)*sR.
__global__ __launch_bounds__(256) void k_ln(const float* __restrict__ in,
                                            const float* __restrict__ g,
                                            const float* __restrict__ be,
                                            float* __restrict__ outF,
                                            bf16* __restrict__ outB,
                                            int M, int P, int sQ, int sR) {
  int row = blockIdx.x * 8 + (threadIdx.x >> 5);
  if (row >= M) return;
  int lane = threadIdx.x & 31;
  size_t ir = (size_t)(row / P) * sQ + (size_t)(row % P) * sR;
  const float* x = in + ir * Dn;
  float v[16], s = 0.f;
#pragma unroll
  for (int i = 0; i < 16; ++i) { v[i] = x[lane + i * 32]; s += v[i]; }
#pragma unroll
  for (int off = 16; off; off >>= 1) s += __shfl_xor(s, off, 32);
  float mean = s * (1.f / 512.f);
  float vs = 0.f;
#pragma unroll
  for (int i = 0; i < 16; ++i) { float d = v[i] - mean; vs += d * d; }
#pragma unroll
  for (int off = 16; off; off >>= 1) vs += __shfl_xor(vs, off, 32);
  float rstd = rsqrtf(vs * (1.f / 512.f) + 1e-5f);
  size_t orow = (size_t)row * Dn;
#pragma unroll
  for (int i = 0; i < 16; ++i) {
    int c = lane + i * 32;
    float y = (v[i] - mean) * rstd * g[c] + be[c];
    if (outF) outF[orow + c] = y;
    if (outB) outB[orow + c] = (bf16)y;
  }
}

// styl: out = bf16(silu( LN(y_row)*(1+scale[b]) + shift[b] )), b = row/T
__global__ __launch_bounds__(256) void k_styl_h(const float* __restrict__ y,
                                                const float* __restrict__ eo,
                                                const float* __restrict__ g,
                                                const float* __restrict__ be,
                                                bf16* __restrict__ out, int M) {
  int row = blockIdx.x * 8 + (threadIdx.x >> 5);
  if (row >= M) return;
  int lane = threadIdx.x & 31;
  int b = row >> 8;  // T=256
  const float* x = y + (size_t)row * Dn;
  float v[16], s = 0.f;
#pragma unroll
  for (int i = 0; i < 16; ++i) { v[i] = x[lane + i * 32]; s += v[i]; }
#pragma unroll
  for (int off = 16; off; off >>= 1) s += __shfl_xor(s, off, 32);
  float mean = s * (1.f / 512.f);
  float vs = 0.f;
#pragma unroll
  for (int i = 0; i < 16; ++i) { float d = v[i] - mean; vs += d * d; }
#pragma unroll
  for (int off = 16; off; off >>= 1) vs += __shfl_xor(vs, off, 32);
  float rstd = rsqrtf(vs * (1.f / 512.f) + 1e-5f);
  size_t orow = (size_t)row * Dn;
#pragma unroll
  for (int i = 0; i < 16; ++i) {
    int c = lane + i * 32;
    float ln = (v[i] - mean) * rstd * g[c] + be[c];
    float sc = eo[(size_t)b * 1024 + c];
    float sh = eo[(size_t)b * 1024 + 512 + c];
    float hv = ln * (1.f + sc) + sh;
    hv = hv / (1.f + __expf(-hv));
    out[orow + c] = (bf16)hv;
  }
}

// ---------------------- linear cross-attention bits ------------------------

__global__ __launch_bounds__(256) void k_softmax_last64(float* __restrict__ x, int nchunks) {
  int chunk = blockIdx.x * 8 + (threadIdx.x >> 5);
  if (chunk >= nchunks) return;
  int lane = threadIdx.x & 31;
  float* p = x + (size_t)chunk * 64;
  float v0 = p[lane], v1 = p[lane + 32];
  float m = fmaxf(v0, v1);
#pragma unroll
  for (int off = 16; off; off >>= 1) m = fmaxf(m, __shfl_xor(m, off, 32));
  float e0 = __expf(v0 - m), e1 = __expf(v1 - m);
  float s = e0 + e1;
#pragma unroll
  for (int off = 16; off; off >>= 1) s += __shfl_xor(s, off, 32);
  float inv = 1.f / s;
  p[lane] = e0 * inv;
  p[lane + 32] = e1 * inv;
}

__global__ void k_softmax_axisN(float* __restrict__ x) {  // softmax over n per (b,c)
  int idx = blockIdx.x * 256 + threadIdx.x;
  if (idx >= Bn * Dn) return;
  int b = idx >> 9, c = idx & 511;
  float m = -1e30f;
  for (int n = 0; n < Nn; ++n) m = fmaxf(m, x[((size_t)(b * Nn + n)) * Dn + c]);
  float s = 0.f;
  for (int n = 0; n < Nn; ++n) s += __expf(x[((size_t)(b * Nn + n)) * Dn + c] - m);
  float inv = 1.f / s;
  for (int n = 0; n < Nn; ++n) {
    size_t o = ((size_t)(b * Nn + n)) * Dn + c;
    x[o] = __expf(x[o] - m) * inv;
  }
}

// ctx[b,h,d,l] = sum_n kc[b,n,h,d] * vc[b,n,h,l]
__global__ __launch_bounds__(256) void k_ctx(const float* __restrict__ kc,
                                             const float* __restrict__ vc,
                                             float* __restrict__ ctx) {
  int bh = blockIdx.x;
  int b = bh >> 3, h = bh & 7;
  __shared__ float sk[64], sv[64];
  float acc[16];
#pragma unroll
  for (int i = 0; i < 16; ++i) acc[i] = 0.f;
  for (int n = 0; n < Nn; ++n) {
    size_t base = ((size_t)(b * Nn + n)) * Dn + h * DHn;
    if (threadIdx.x < 64)       sk[threadIdx.x] = kc[base + threadIdx.x];
    else if (threadIdx.x < 128) sv[threadIdx.x - 64] = vc[base + threadIdx.x - 64];
    __syncthreads();
#pragma unroll
    for (int i = 0; i < 16; ++i) {
      int idx = threadIdx.x + i * 256;
      acc[i] += sk[idx >> 6] * sv[idx & 63];
    }
    __syncthreads();
  }
  size_t ob = (size_t)bh * 4096;
#pragma unroll
  for (int i = 0; i < 16; ++i) ctx[ob + threadIdx.x + i * 256] = acc[i];
}

// y[b,t,h,l] = sum_d qc[b,t,h,d] * ctx[b,h,d,l]
__global__ __launch_bounds__(256) void k_y(const float* __restrict__ qc,
                                           const float* __restrict__ ctx,
                                           float* __restrict__ y) {
  int row = blockIdx.x;      // b*T + t
  int b = row >> 8;
  __shared__ float sq[512];
  sq[threadIdx.x] = qc[(size_t)row * Dn + threadIdx.x];
  sq[threadIdx.x + 256] = qc[(size_t)row * Dn + threadIdx.x + 256];
  __syncthreads();
#pragma unroll
  for (int i = 0; i < 2; ++i) {
    int c = threadIdx.x + i * 256;
    int h = c >> 6, l = c & 63;
    const float* cp = ctx + ((size_t)(b * 8 + h)) * 4096 + l;
    float s = 0.f;
#pragma unroll 8
    for (int d = 0; d < 64; ++d) s += sq[h * 64 + d] * cp[d * 64];
    y[(size_t)row * Dn + c] = s;
  }
}

// ------------------------------- launch ------------------------------------

extern "C" void kernel_launch(void* const* d_in, const int* in_sizes, int n_in,
                              void* d_out, int out_size, void* d_ws, size_t ws_size,
                              hipStream_t stream) {
  (void)in_sizes; (void)n_in; (void)out_size; (void)ws_size;
  const float* x    = (const float*)d_in[0];
  const float* xf   = (const float*)d_in[1];
  const float* emb  = (const float*)d_in[2];
  const float* sa_in_b  = (const float*)d_in[4];
  const float* sa_out_b = (const float*)d_in[6];
  const float* sa_l1_b  = (const float*)d_in[8];
  const float* sa_l2_b  = (const float*)d_in[10];
  const float* sa_n1_g  = (const float*)d_in[11];
  const float* sa_n1_b  = (const float*)d_in[12];
  const float* sa_n2_g  = (const float*)d_in[13];
  const float* sa_n2_b  = (const float*)d_in[14];
  const float* ca_norm_g  = (const float*)d_in[15];
  const float* ca_norm_b  = (const float*)d_in[16];
  const float* ca_tnorm_g = (const float*)d_in[17];
  const float* ca_tnorm_b = (const float*)d_in[18];
  const float* ca_q_b  = (const float*)d_in[20];
  const float* ca_k_b  = (const float*)d_in[22];
  const float* ca_v_b  = (const float*)d_in[24];
  const float* ca_se_b = (const float*)d_in[26];
  const float* ca_sn_g = (const float*)d_in[27];
  const float* ca_sn_b = (const float*)d_in[28];
  const float* ca_so_b = (const float*)d_in[30];
  const float* ff_l1_b = (const float*)d_in[32];
  const float* ff_l2_b = (const float*)d_in[34];
  const float* ff_se_b = (const float*)d_in[36];
  const float* ff_sn_g = (const float*)d_in[37];
  const float* ff_sn_b = (const float*)d_in[38];
  const float* ff_so_b = (const float*)d_in[40];

  char* ws = (char*)d_ws;
  auto al = [](size_t v) { return (v + 255) & ~(size_t)255; };

  // --- regions (total ~344 MB) ---
  const size_t o_w    = 0;                               // bf16 weight pool (13.1 MB)
  const size_t sz_w   = (size_t)6553600 * 2;
  const size_t o_srcF = al(o_w + sz_w);                  // 43.8 MB  (later: res2, xb2_bf16)
  const size_t o_srcB = al(o_srcF + SBD * 4);            // 21.9 MB  (later: xn, h, h2)
  const size_t o_R3   = al(o_srcB + SBD * 2);            // 131.3 MB arena
  const size_t o_R4   = al(o_R3 + SBD * 12);             // 21.9 MB  (o_bf16; later tn+kraw)
  const size_t o_R5   = al(o_R4 + SBD * 2);              // 43.8 MB  (mid1; later qraw/qc)
  const size_t o_xb   = al(o_R5 + SBD * 4);              // 33.6 MB  (xb; later out tmp)
  const size_t o_xb2  = al(o_xb + BTD * 4);              // 33.6 MB  (xb2 f32)
  const size_t o_sm   = al(o_xb2 + BTD * 4);             // small: silu(e), eo, eo2

  bf16*  wpool = (bf16*)(ws + o_w);
  float* srcF  = (float*)(ws + o_srcF);
  bf16*  srcB  = (bf16*)(ws + o_srcB);

  float* qkvF  = (float*)(ws + o_R3);                    // [SB,1536]
  float* res1F = (float*)(ws + o_R3);                    // alias (qkv dead)
  float* src1F = (float*)(ws + o_R3 + SBD * 4);
  bf16*  src1B = (bf16*)(ws + o_R3 + SBD * 8);
  float* src2F = (float*)(ws + o_R3);                    // alias (res1 dead)
  float* vcF   = (float*)(ws + o_R3 + SBD * 4);          // alias (src1 dead)
  float* ctxF  = (float*)(ws + o_R3 + SBD * 8);
  float* yF    = (float*)(ws + o_R3 + SBD * 8 + (size_t)8388608);
  bf16*  mid2B = (bf16*)(ws + o_R3);                     // alias (all above dead)
  float* y2F   = (float*)(ws + o_R3 + (size_t)67108864);

  bf16*  oB    = (bf16*)(ws + o_R4);
  bf16*  tnB   = (bf16*)(ws + o_R4);                     // alias (o dead)
  float* krawF = (float*)(ws + o_R4 + BND * 2);

  bf16*  mid1B = (bf16*)(ws + o_R5);
  float* qrawF = (float*)(ws + o_R5);                    // alias (mid1 dead)

  bf16*  xnB   = (bf16*)(ws + o_srcB);                   // alias (srcB dead)
  bf16*  hB    = (bf16*)(ws + o_srcB);                   // alias (xn dead)
  bf16*  h2B   = (bf16*)(ws + o_srcB);                   // alias (h dead)
  float* res2F = (float*)(ws + o_srcF);                  // alias (srcF dead)
  bf16*  xb2B  = (bf16*)(ws + o_srcF);                   // alias (res2 dead)

  float* xbF   = (float*)(ws + o_xb);
  float* outT  = (float*)(ws + o_xb);                    // alias (xb dead)
  float* xb2F  = (float*)(ws + o_xb2);

  bf16*  seB   = (bf16*)(ws + o_sm);
  float* eoF   = (float*)(ws + al(o_sm + (size_t)Bn * Dn * 2));
  float* eo2F  = (float*)(ws + al(o_sm + (size_t)Bn * Dn * 2) + al((size_t)Bn * 1024 * 4));

  // --- 1. weights -> bf16 pool ---
  struct WEnt { int idx; size_t n; size_t off; };
  const WEnt wl[13] = {
      {3, 786432, 0},        {5, 262144, 786432},   {7, 524288, 1048576},
      {9, 524288, 1572864},  {19, 262144, 2097152}, {21, 262144, 2359296},
      {23, 262144, 2621440}, {25, 524288, 2883584}, {29, 262144, 3407872},
      {31, 1048576, 3670016},{33, 1048576, 4718592},{35, 524288, 5767168},
      {39, 262144, 6291456}};
  for (int i = 0; i < 13; ++i)
    k_f2b<<<dim3((unsigned)((wl[i].n + 255) / 256)), 256, 0, stream>>>(
        (const float*)d_in[wl[i].idx], wpool + wl[i].off, wl[i].n);
  const bf16* wb_sa_in = wpool + 0;
  const bf16* wb_sa_out = wpool + 786432;
  const bf16* wb_sa_l1 = wpool + 1048576;
  const bf16* wb_sa_l2 = wpool + 1572864;
  const bf16* wb_ca_q = wpool + 2097152;
  const bf16* wb_ca_k = wpool + 2359296;
  const bf16* wb_ca_v = wpool + 2621440;
  const bf16* wb_ca_se = wpool + 2883584;
  const bf16* wb_ca_so = wpool + 3407872;
  const bf16* wb_ff_l1 = wpool + 3670016;
  const bf16* wb_ff_l2 = wpool + 4718592;
  const bf16* wb_ff_se = wpool + 5767168;
  const bf16* wb_ff_so = wpool + 6291456;

  auto gemm = [&](const bf16* A, const bf16* W, const float* bias, const float* resid,
                  float* oF, bf16* oBp, int M, int N, int K, int act) {
    dim3 g((unsigned)(N / GBN), (unsigned)((M + GBM - 1) / GBM));
    k_gemm<<<g, 256, 0, stream>>>(A, W, bias, resid, oF, oBp, M, N, K, act);
  };
  const int BIGP = 1 << 30;

  // --- 2. concat src ---
  k_build_src<<<dim3((unsigned)((SBD + 255) / 256)), 256, 0, stream>>>(x, xf, emb, srcF, srcB);

  // --- 3..9: self-attention block + FFN + LNs ---
  gemm(srcB, wb_sa_in, sa_in_b, nullptr, qkvF, nullptr, SB, 1536, 512, 0);
  k_attn<<<dim3(3, Bn * Hn), 256, 0, stream>>>(qkvF, oB);
  gemm(oB, wb_sa_out, sa_out_b, srcF, res1F, nullptr, SB, 512, 512, 0);
  k_ln<<<dim3((SB + 7) / 8), 256, 0, stream>>>(res1F, sa_n1_g, sa_n1_b, src1F, src1B,
                                               SB, BIGP, 0, 1);
  gemm(src1B, wb_sa_l1, sa_l1_b, nullptr, nullptr, mid1B, SB, 1024, 512, 1);
  gemm(mid1B, wb_sa_l2, sa_l2_b, src1F, res2F, nullptr, SB, 512, 1024, 0);
  k_ln<<<dim3((SB + 7) / 8), 256, 0, stream>>>(res2F, sa_n2_g, sa_n2_b, src2F, nullptr,
                                               SB, BIGP, 0, 1);

  // --- 10..19: linear temporal cross-attention ---
  k_permute<<<dim3((unsigned)((BTD + 255) / 256)), 256, 0, stream>>>(src2F, xbF, BT, Tn, 1, Bn);
  k_ln<<<dim3((BT + 7) / 8), 256, 0, stream>>>(xbF, ca_norm_g, ca_norm_b, nullptr, xnB,
                                               BT, BIGP, 0, 1);
  k_ln<<<dim3((BNr + 7) / 8), 256, 0, stream>>>(xf, ca_tnorm_g, ca_tnorm_b, nullptr, tnB,
                                                BNr, Nn, 1, Bn);
  gemm(xnB, wb_ca_q, ca_q_b, nullptr, qrawF, nullptr, BT, 512, 512, 0);
  gemm(tnB, wb_ca_k, ca_k_b, nullptr, krawF, nullptr, BNr, 512, 512, 0);
  gemm(tnB, wb_ca_v, ca_v_b, nullptr, vcF, nullptr, BNr, 512, 512, 0);
  k_softmax_last64<<<dim3((BT * Hn + 7) / 8), 256, 0, stream>>>(qrawF, BT * Hn);
  k_softmax_axisN<<<dim3((Bn * Dn + 255) / 256), 256, 0, stream>>>(krawF);
  k_ctx<<<dim3(Bn * Hn), 256, 0, stream>>>(krawF, vcF, ctxF);
  k_y<<<dim3(BT), 256, 0, stream>>>(qrawF, ctxF, yF);

  // stylization (CA)
  k_silu_b<<<dim3((Bn * Dn + 255) / 256), 256, 0, stream>>>(emb, seB, (size_t)Bn * Dn);
  gemm(seB, wb_ca_se, ca_se_b, nullptr, eoF, nullptr, Bn, 1024, 512, 0);
  k_styl_h<<<dim3((BT + 7) / 8), 256, 0, stream>>>(yF, eoF, ca_sn_g, ca_sn_b, hB, BT);
  gemm(hB, wb_ca_so, ca_so_b, xbF, xb2F, xb2B, BT, 512, 512, 0);

  // --- 20..28: FFN with stylization ---
  gemm(xb2B, wb_ff_l1, ff_l1_b, nullptr, nullptr, mid2B, BT, 2048, 512, 2);
  gemm(mid2B, wb_ff_l2, ff_l2_b, nullptr, y2F, nullptr, BT, 512, 2048, 0);
  gemm(seB, wb_ff_se, ff_se_b, nullptr, eo2F, nullptr, Bn, 1024, 512, 0);
  k_styl_h<<<dim3((BT + 7) / 8), 256, 0, stream>>>(y2F, eo2F, ff_sn_g, ff_sn_b, h2B, BT);
  gemm(h2B, wb_ff_so, ff_so_b, xb2F, outT, nullptr, BT, 512, 512, 0);

  // --- 29: [B,T,D] -> [T,B,D] output ---
  k_permute<<<dim3((unsigned)((BTD + 255) / 256)), 256, 0, stream>>>(outT, (float*)d_out,
                                                                    Tn * Bn, Bn, 1, Tn);
}
